// DifferentialAttention_26371099197741
// MI455X (gfx1250) — matrix operand order repaired and checked by reading, not compiled, not run on hardware
//
#include <hip/hip_runtime.h>
#include <hip/hip_bf16.h>
#include <math.h>

// ---------------- CDNA5 WMMA helpers (gfx1250, wave32) ----------------
typedef __attribute__((ext_vector_type(16))) _Float16 v16h;
typedef __attribute__((ext_vector_type(8)))  _Float16 v8h;
typedef __attribute__((ext_vector_type(2)))  _Float16 v2h;
typedef __attribute__((ext_vector_type(8)))  float    v8f;

static __device__ __forceinline__ v8f wmma_(v16h a, v16h b, v8f c) {
  // D = A(16x32 f16) x B(32x16 f16) + C(16x16 f32)
  return __builtin_amdgcn_wmma_f32_16x16x32_f16(false, a, false, b, (short)0, c, false, false);
}

// A-operand load: per lane, 8 halves at p and 8 halves at p+16 (K-dim layout:
// elems 0..7 = K(hi*8+0..7), elems 8..15 = K(16+hi*8+0..7))
static __device__ __forceinline__ v16h ld_a(const _Float16* p) {
  v8h lo = *(const v8h*)p;
  v8h hv = *(const v8h*)(p + 16);
  return __builtin_shufflevector(lo, hv, 0,1,2,3,4,5,6,7,8,9,10,11,12,13,14,15);
}

static __device__ __forceinline__ int iabs_(int x) { return x < 0 ? -x : x; }

// ---------------- problem constants ----------------
constexpr int BATCH = 4;
constexpr int SEQ   = 1024;   // 32x32 grid
constexpr int EMB   = 768;
constexpr int NH    = 12;
constexpr int NKV   = 4;
constexpr int HD    = 64;
constexpr int SIDE  = 32;
constexpr int WIN   = 8;
constexpr int ROWS  = BATCH * SEQ; // 4096

// ---------------- f32 -> f16 convert ----------------
__global__ void cvt_kernel(const float* __restrict__ in, _Float16* __restrict__ out, int n) {
  int i = blockIdx.x * blockDim.x + threadIdx.x;
  if (i < n) out[i] = (_Float16)in[i];
}

// ---------------- generic GEMM: out(MxNN,f32) = A(MxK,f16) @ W(NNxK,f16)^T (+bias) ----
// block = 256 threads = 8 waves arranged 2(M) x 4(N); wave tile = 32(M) x 64(N);
// block tile = 64 x 256; 8 WMMAs per 32-k step, 12 b128 loads -> 1.5 loads/WMMA.
__global__ __launch_bounds__(256) void gemm_kernel(const _Float16* __restrict__ A,
                                                   const _Float16* __restrict__ W,
                                                   const float* __restrict__ bias,
                                                   float* __restrict__ out,
                                                   int Kdim, int NN) {
  const int w    = threadIdx.x >> 5;
  const int lane = threadIdx.x & 31;
  const int lq   = lane & 15;
  const int hi   = lane >> 4;
  const int wm   = w >> 2;      // 0..1
  const int wn   = w & 3;       // 0..3
  const int m0   = blockIdx.y * 64 + wm * 32;
  const int n0   = blockIdx.x * 256 + wn * 64;

  const _Float16* arow0 = A + (size_t)(m0 + lq) * Kdim + hi * 8;
  const _Float16* arow1 = A + (size_t)(m0 + 16 + lq) * Kdim + hi * 8;
  const _Float16* wp[4];
  #pragma unroll
  for (int j = 0; j < 4; ++j)
    wp[j] = W + (size_t)(n0 + j * 16 + lq) * Kdim + hi * 8;

  v8f acc[2][4];
  v8f zv = {};
  #pragma unroll
  for (int i = 0; i < 2; ++i)
    #pragma unroll
    for (int j = 0; j < 4; ++j) acc[i][j] = zv;

  #pragma unroll 2
  for (int k0 = 0; k0 < Kdim; k0 += 32) {
    v16h a0 = ld_a(arow0 + k0);
    v16h a1 = ld_a(arow1 + k0);
    v16h b0 = ld_a(wp[0] + k0);
    v16h b1 = ld_a(wp[1] + k0);
    v16h b2 = ld_a(wp[2] + k0);
    v16h b3 = ld_a(wp[3] + k0);
    acc[0][0] = wmma_(a0, b0, acc[0][0]);
    acc[0][1] = wmma_(a0, b1, acc[0][1]);
    acc[0][2] = wmma_(a0, b2, acc[0][2]);
    acc[0][3] = wmma_(a0, b3, acc[0][3]);
    acc[1][0] = wmma_(a1, b0, acc[1][0]);
    acc[1][1] = wmma_(a1, b1, acc[1][1]);
    acc[1][2] = wmma_(a1, b2, acc[1][2]);
    acc[1][3] = wmma_(a1, b3, acc[1][3]);
  }

  #pragma unroll
  for (int j = 0; j < 4; ++j) {
    float bb = bias ? bias[n0 + j * 16 + lq] : 0.f;
    #pragma unroll
    for (int i = 0; i < 2; ++i) {
      #pragma unroll
      for (int r = 0; r < 8; ++r) {
        out[(size_t)(m0 + i * 16 + r + 8 * hi) * NN + n0 + j * 16 + lq] = acc[i][j][r] + bb;
      }
    }
  }
}

// ---------------- RoPE (2D) + qk-norm + layout epilogue ----------------
// one wave (32 lanes) per 64-dim head-row; lane i owns rope pair (2i, 2i+1)
__global__ __launch_bounds__(256) void rope_norm_kernel(const float* __restrict__ projq,
                                                        const float* __restrict__ projk,
                                                        const float* __restrict__ projv1,
                                                        const float* __restrict__ projv2,
                                                        _Float16* __restrict__ qbuf,
                                                        _Float16* __restrict__ kbuf,
                                                        _Float16* __restrict__ vtbuf) {
  const int lane = threadIdx.x & 31;
  const int gw   = blockIdx.x * (blockDim.x >> 5) + (threadIdx.x >> 5);
  const int QR = BATCH * SEQ * 2 * NH;   // 98304
  const int KR = BATCH * SEQ * 2 * NKV;  // 32768
  const int VR = BATCH * SEQ * 2 * NKV;  // 32768
  if (gw >= QR + KR + VR) return;

  if (gw < QR + KR) {
    // --- q or k row: rope + norm ---
    int t, hh, e, n, b;
    const float* src;
    _Float16* dst;
    if (gw < QR) {
      t = gw; hh = t % NH; t /= NH; e = t & 1; t >>= 1; n = t % SEQ; b = t / SEQ;
      src = projq + ((size_t)b * SEQ + n) * (2 * EMB) + e * EMB + hh * HD;
      dst = qbuf + (((size_t)((e * BATCH + b) * NH + hh)) * SEQ + n) * HD;
    } else {
      t = gw - QR; hh = t % NKV; t /= NKV; e = t & 1; t >>= 1; n = t % SEQ; b = t / SEQ;
      src = projk + ((size_t)b * SEQ + n) * 512 + e * 256 + hh * HD;
      dst = kbuf + (((size_t)((e * BATCH + b) * NKV + hh)) * SEQ + n) * HD;
    }
    float x0 = src[2 * lane], x1 = src[2 * lane + 1];
    float ph = (float)(n >> 5), pw = (float)(n & 31);
    int j0 = 2 * lane;
    float pos = (j0 < 32) ? ph : pw;
    float invf0 = powf(10000.f, -(float)(j0 & 31)       * (1.f / 32.f));
    float invf1 = powf(10000.f, -(float)((j0 + 1) & 31) * (1.f / 32.f));
    float a0 = pos * invf0, a1 = pos * invf1;
    float c0 = cosf(a0), s0 = sinf(a0), c1 = cosf(a1), s1 = sinf(a1);
    float r0 = x0 * c0 - x1 * s0;     // even dim: q*cos - q_odd*sin
    float r1 = x1 * c1 + x0 * s1;     // odd  dim: q*cos + q_even*sin
    float ss = r0 * r0 + r1 * r1;
    #pragma unroll
    for (int o = 16; o > 0; o >>= 1) ss += __shfl_xor(ss, o, 32);
    float sc = 1.f / (sqrtf(ss) + 1e-6f);
    v2h pr; pr[0] = (_Float16)(r0 * sc); pr[1] = (_Float16)(r1 * sc);
    *(v2h*)(dst + 2 * lane) = pr;
  } else {
    // --- v row: convert + transposed store vT[e][b][kv][d][n] ---
    int t = gw - QR - KR;
    int kv = t % NKV; t /= NKV; int e = t & 1; t >>= 1; int n = t % SEQ; int b = t / SEQ;
    const float* src = (e == 0 ? projv1 : projv2) + ((size_t)b * SEQ + n) * 256 + kv * HD;
    _Float16* base = vtbuf + ((size_t)((e * BATCH + b) * NKV + kv)) * HD * SEQ;
    float x0 = src[2 * lane], x1 = src[2 * lane + 1];
    base[(size_t)(2 * lane) * SEQ + n]     = (_Float16)x0;
    base[(size_t)(2 * lane + 1) * SEQ + n] = (_Float16)x1;
  }
}

// ---------------- differential windowed flash attention ----------------
// one wave per (b, h, 16-query tile); both branches; out = o1 - softplus(lam)*o2
__global__ __launch_bounds__(256) void attn_kernel(const _Float16* __restrict__ qbuf,
                                                   const _Float16* __restrict__ kbuf,
                                                   const _Float16* __restrict__ vtbuf,
                                                   const float* __restrict__ lambda_p,
                                                   _Float16* __restrict__ attnout) {
  __shared__ __attribute__((aligned(64))) _Float16 plds[8][16][32]; // per-wave P^T staging
  const int w    = threadIdx.x >> 5;
  const int lane = threadIdx.x & 31;
  const int lq   = lane & 15;
  const int hi   = lane >> 4;

  const int gwave = blockIdx.x * 8 + w;       // 0..3071
  const int qt = gwave & 63;                  // 16-query tile in sequence
  const int h  = (gwave >> 6) % NH;
  const int b  = gwave / (64 * NH);
  const int kv = h / (NH / NKV);

  const int qh = qt >> 1;                     // grid row of all 16 queries
  const int r0 = (qh - WIN < 0) ? 0 : qh - WIN;
  const int r1 = (qh + WIN > SIDE - 1) ? SIDE - 1 : qh + WIN;
  const int qw = ((qt & 1) << 4) + lq;        // this lane's query grid-col

  // column window mask (step-invariant): key-in-step = c*16 + hi*8 + r == kw
  v8f mk0, mk1;
  #pragma unroll
  for (int r = 0; r < 8; ++r) {
    int kw0 = hi * 8 + r, kw1 = 16 + hi * 8 + r;
    mk0[r] = (iabs_(qw - kw0) <= WIN) ? 0.f : -1e30f;
    mk1[r] = (iabs_(qw - kw1) <= WIN) ? 0.f : -1e30f;
  }
  const float lam = log1pf(expf(lambda_p[h]));
  const float scale = 0.125f; // 1/sqrt(64)

  v8f zv = {};
  v8f o[4];  v8f o1[4];
  #pragma unroll
  for (int t = 0; t < 4; ++t) { o[t] = zv; o1[t] = zv; }

  for (int e = 0; e < 2; ++e) {
    const size_t kvi = (size_t)((e * BATCH + b) * NKV + kv);
    const _Float16* Kb = kbuf  + kvi * (size_t)SEQ * HD;
    const _Float16* Vt = vtbuf + kvi * (size_t)HD * SEQ;
    const _Float16* Qb = qbuf + ((size_t)((e * BATCH + b) * NH + h)) * SEQ * HD
                        + (size_t)qt * 16 * HD;
    // Q^T B-operands (col = query lq, K-dim = d)
    v16h bq0 = ld_a(Qb + lq * HD + hi * 8);
    v16h bq1 = ld_a(Qb + lq * HD + 32 + hi * 8);

    float m = -1e30f, l = 0.f;
    #pragma unroll
    for (int t = 0; t < 4; ++t) o[t] = zv;

    for (int s = r0; s <= r1; ++s) {
      const int jb = s << 5;  // 32 keys = one full grid row
      // S^T = K . Q^T : two 16x16 tiles (keys 0-15, 16-31), 2 d-chunks each
      const _Float16* kr0 = Kb + (size_t)(jb + lq) * HD + hi * 8;
      const _Float16* kr1 = Kb + (size_t)(jb + 16 + lq) * HD + hi * 8;
      v8f st0 = zv, st1 = zv;
      st0 = wmma_(ld_a(kr0),      bq0, st0);
      st0 = wmma_(ld_a(kr0 + 32), bq1, st0);
      st1 = wmma_(ld_a(kr1),      bq0, st1);
      st1 = wmma_(ld_a(kr1 + 32), bq1, st1);

      // scale + mask + online softmax (key-reduction is in-lane + one shfl)
      float pv[16]; float mloc = -1e30f;
      #pragma unroll
      for (int r = 0; r < 8; ++r) {
        float v0 = st0[r] * scale + mk0[r];
        float v1 = st1[r] * scale + mk1[r];
        pv[r] = v0; pv[8 + r] = v1;
        mloc = fmaxf(mloc, fmaxf(v0, v1));
      }
      mloc = fmaxf(mloc, __shfl_xor(mloc, 16, 32));
      float mnew = fmaxf(m, mloc);
      float corr = expf(m - mnew);
      m = mnew; l *= corr;
      #pragma unroll
      for (int t = 0; t < 4; ++t) o[t] *= corr;

      float ls = 0.f;
      v8h ph0, ph1;
      #pragma unroll
      for (int r = 0; r < 8; ++r) {
        float p0 = expf(pv[r] - m);
        float p1 = expf(pv[8 + r] - m);
        ls += p0 + p1;
        ph0[r] = (_Float16)p0; ph1[r] = (_Float16)p1;
      }
      ls += __shfl_xor(ls, 16, 32);
      l += ls;

      // pack P^T -> LDS [q][key]: consecutive VGPRs are consecutive keys
      *(v8h*)&plds[w][lq][hi * 8]      = ph0;
      *(v8h*)&plds[w][lq][16 + hi * 8] = ph1;
      __asm__ volatile("" ::: "memory");
      v16h bP = ld_a(&plds[w][lq][hi * 8]); // B-operand (32 keys x 16 q)
      __asm__ volatile("" ::: "memory");

      // O^T += V^T . P^T  (A = V^T tile, contiguous from transposed V buffer)
      #pragma unroll
      for (int t = 0; t < 4; ++t) {
        const _Float16* vp = Vt + (size_t)(t * 16 + lq) * SEQ + jb + hi * 8;
        o[t] = wmma_(ld_a(vp), bP, o[t]);
      }
    }
    float linv = 1.f / l;
    #pragma unroll
    for (int t = 0; t < 4; ++t) o[t] *= linv;
    if (e == 0) {
      #pragma unroll
      for (int t = 0; t < 4; ++t) o1[t] = o[t];
    }
  }

  // combine branches and store f16 into [b, n, h*64+d] for the Wo GEMM
  _Float16* orow = attnout + ((size_t)b * SEQ + qt * 16 + lq) * EMB + h * HD;
  #pragma unroll
  for (int t = 0; t < 4; ++t) {
    v8h ph;
    #pragma unroll
    for (int r = 0; r < 8; ++r) ph[r] = (_Float16)(o1[t][r] - lam * o[t][r]);
    *(v8h*)(orow + t * 16 + hi * 8) = ph;
  }
}

// ---------------- launcher ----------------
extern "C" void kernel_launch(void* const* d_in, const int* in_sizes, int n_in,
                              void* d_out, int out_size, void* d_ws, size_t ws_size,
                              hipStream_t stream) {
  (void)in_sizes; (void)n_in; (void)out_size; (void)ws_size;
  const float* x   = (const float*)d_in[0];
  const float* Wq  = (const float*)d_in[1];
  const float* Wk  = (const float*)d_in[2];
  const float* Wv1 = (const float*)d_in[3];
  const float* Wv2 = (const float*)d_in[4];
  const float* lp  = (const float*)d_in[5];
  const float* Wo  = (const float*)d_in[6];
  const float* bo  = (const float*)d_in[7];
  float* out = (float*)d_out;

  char* p = (char*)d_ws;
  _Float16* xh    = (_Float16*)p; p += (size_t)ROWS * EMB * 2;        // 6.29 MB
  _Float16* Wqh   = (_Float16*)p; p += (size_t)2 * EMB * EMB * 2;     // 2.36 MB
  _Float16* Wkh   = (_Float16*)p; p += (size_t)512 * EMB * 2;
  _Float16* Wv1h  = (_Float16*)p; p += (size_t)256 * EMB * 2;
  _Float16* Wv2h  = (_Float16*)p; p += (size_t)256 * EMB * 2;
  _Float16* Woh   = (_Float16*)p; p += (size_t)EMB * EMB * 2;
  float* projq  = (float*)p; p += (size_t)ROWS * 1536 * 4;            // 25.2 MB
  float* projk  = (float*)p; p += (size_t)ROWS * 512 * 4;
  float* projv1 = (float*)p; p += (size_t)ROWS * 256 * 4;
  float* projv2 = (float*)p; p += (size_t)ROWS * 256 * 4;
  _Float16* qbuf  = (_Float16*)p; p += (size_t)2 * BATCH * NH  * SEQ * HD * 2;
  _Float16* kbuf  = (_Float16*)p; p += (size_t)2 * BATCH * NKV * SEQ * HD * 2;
  _Float16* vtbuf = (_Float16*)p; p += (size_t)2 * BATCH * NKV * HD * SEQ * 2;
  _Float16* attnb = (_Float16*)p; p += (size_t)ROWS * EMB * 2;

  auto cvt = [&](const float* s, _Float16* d, int n) {
    cvt_kernel<<<(n + 255) / 256, 256, 0, stream>>>(s, d, n);
  };
  cvt(x,   xh,   ROWS * EMB);
  cvt(Wq,  Wqh,  2 * EMB * EMB);
  cvt(Wk,  Wkh,  512 * EMB);
  cvt(Wv1, Wv1h, 256 * EMB);
  cvt(Wv2, Wv2h, 256 * EMB);
  cvt(Wo,  Woh,  EMB * EMB);

  // projections: out(4096 x NN) = xh @ W^T ; block tile 64(M) x 256(N)
  gemm_kernel<<<dim3(6, ROWS / 64), 256, 0, stream>>>(xh, Wqh,  nullptr, projq,  EMB, 1536);
  gemm_kernel<<<dim3(2, ROWS / 64), 256, 0, stream>>>(xh, Wkh,  nullptr, projk,  EMB, 512);
  gemm_kernel<<<dim3(1, ROWS / 64), 256, 0, stream>>>(xh, Wv1h, nullptr, projv1, EMB, 256);
  gemm_kernel<<<dim3(1, ROWS / 64), 256, 0, stream>>>(xh, Wv2h, nullptr, projv2, EMB, 256);

  // rope + qk-norm + layout: 163840 head-rows, 8 waves/block
  rope_norm_kernel<<<20480, 256, 0, stream>>>(projq, projk, projv1, projv2, qbuf, kbuf, vtbuf);

  // attention: 3072 wave-tiles, 8 waves/block
  attn_kernel<<<384, 256, 0, stream>>>(qbuf, kbuf, vtbuf, lp, attnb);

  // output projection with fused bias
  gemm_kernel<<<dim3(3, ROWS / 64), 256, 0, stream>>>(attnb, Woh, bo, out, EMB, EMB);
}
